// MAXISLoss_18769007084526
// MI455X (gfx1250) — matrix operand order repaired and checked by reading, not compiled, not run on hardware
//
#include <hip/hip_runtime.h>
#include <hip/hip_bf16.h>
#include <math.h>

#define VOCAB  32000
#define DMODEL 1024
#define LOW    64
#define NCAND  2048
#define CHUNK  32
#define STRIDE 4
#define NTOK   2048
#define CCH    (NTOK / CHUNK)      // 64 chunks
#define NS     (CHUNK / STRIDE)    // 8 scouts per chunk
#define KPER   (NCAND / NS)        // 256 candidates per scout
#define MCAND  (NS * KPER)         // 2048 candidates per chunk
#define NTILES (MCAND / 16)        // 128 N tiles of 16
#define NSPLIT 4                   // candidate-dim splits per chunk
#define LTILES (NTILES / NSPLIT)   // 32 local tiles per block
#define VREM   (VOCAB - MCAND - 1) // 29951
#define ASTRIDE 1032               // padded LDS row stride (ushort), 16B-aligned rows
#define SSTRIDE 72                 // scout A-tile LDS row stride (ushort)

typedef __attribute__((ext_vector_type(16))) __bf16 v16bf;
typedef __attribute__((ext_vector_type(8)))  float  v8f;

__device__ __forceinline__ unsigned short f2bf(float x) {
  unsigned u = __float_as_uint(x);
  unsigned r = u + 0x7FFFu + ((u >> 16) & 1u);
  return (unsigned short)(r >> 16);
}

__device__ __forceinline__ unsigned f2key(float x) {
  unsigned u = __float_as_uint(x);
  return (u & 0x80000000u) ? ~u : (u | 0x80000000u);
}

// B fragment (K x N = 32x16): lane = column n; elements = K off..off+15 contiguous
__device__ __forceinline__ v16bf loadB(const unsigned short* wbf, const float* w,
                                       int cid, int off) {
  union { uint4 q[2]; v16bf v; } r;
  if (wbf) {
    const unsigned short* p = wbf + (size_t)cid * DMODEL + off;
    r.q[0] = *reinterpret_cast<const uint4*>(p);
    r.q[1] = *reinterpret_cast<const uint4*>(p + 8);
  } else {
    const float4* p = reinterpret_cast<const float4*>(w + (size_t)cid * DMODEL + off);
    union { unsigned short u[16]; uint4 q[2]; } t16;
#pragma unroll
    for (int j = 0; j < 4; j++) {
      float4 t = p[j];
      t16.u[j * 4 + 0] = f2bf(t.x); t16.u[j * 4 + 1] = f2bf(t.y);
      t16.u[j * 4 + 2] = f2bf(t.z); t16.u[j * 4 + 3] = f2bf(t.w);
    }
    r.q[0] = t16.q[0]; r.q[1] = t16.q[1];
  }
  return r.v;
}

// A fragment (M x K = 16x32): lane = row; K packing h*16+j contiguous
__device__ __forceinline__ v16bf loadA(const unsigned short* row, int kb, int lhalf) {
  union { uint4 q[2]; v16bf v; } r;
  r.q[0] = *reinterpret_cast<const uint4*>(row + kb + lhalf * 16);
  r.q[1] = *reinterpret_cast<const uint4*>(row + kb + lhalf * 16 + 8);
  return r.v;
}

// ---------------------------------------------------------------------------
__global__ void k_norms(const float* __restrict__ w, float* __restrict__ ws_f) {
  const int blk  = blockIdx.x;      // 128 blocks
  const int tid  = threadIdx.x;     // 256 threads = 8 waves
  const int wave = tid >> 5, lane = tid & 31;
  __shared__ float part[8][2];
  float sf = 0.f, sl = 0.f;
  const int rows_per = VOCAB / 128; // 250
  const int r0 = blk * rows_per;
  for (int r = r0 + wave; r < r0 + rows_per; r += 8) {
    const float* row = w + (size_t)r * DMODEL;
    float a = 0.f, b = 0.f;
    for (int k = lane; k < DMODEL; k += 32) {
      float v = row[k];
      a += v * v;
      if (k < LOW) b += v * v;
    }
    sf += a; sl += b;
  }
  for (int m = 16; m; m >>= 1) { sf += __shfl_xor(sf, m, 32); sl += __shfl_xor(sl, m, 32); }
  if (lane == 0) { part[wave][0] = sf; part[wave][1] = sl; }
  __syncthreads();
  if (tid == 0) {
    float a = 0.f, b = 0.f;
    for (int i = 0; i < 8; i++) { a += part[i][0]; b += part[i][1]; }
    ws_f[4 + blk * 2]     = a;
    ws_f[4 + blk * 2 + 1] = b;
  }
}

__global__ void k_norm_combine(float* __restrict__ ws_f) {
  if (threadIdx.x == 0 && blockIdx.x == 0) {
    float a = 0.f, b = 0.f;
    for (int i = 0; i < 128; i++) { a += ws_f[4 + i * 2]; b += ws_f[4 + i * 2 + 1]; }
    ws_f[0] = a;
    ws_f[1] = b;
  }
}

// ---------------------------------------------------------------------------
// wave-per-token: pos logits and ||h||^2 for both branches
// ---------------------------------------------------------------------------
__global__ void k_pos(const float* __restrict__ hidden, const float* __restrict__ w,
                      const int* __restrict__ tgt, float* __restrict__ posbuf) {
  const int tok  = blockIdx.x * 8 + (threadIdx.x >> 5);
  const int lane = threadIdx.x & 31;
  const float* hrow = hidden + (size_t)tok * DMODEL;
  const float* wrow = w + (size_t)tgt[tok] * DMODEL;
  float pf = 0.f, hf = 0.f, pl = 0.f, hl = 0.f;
  for (int k = lane; k < DMODEL; k += 32) {
    float hv = hrow[k], wv = wrow[k];
    pf += hv * wv; hf += hv * hv;
    if (k < LOW) { pl += hv * wv; hl += hv * hv; }
  }
  for (int m = 16; m; m >>= 1) {
    pf += __shfl_xor(pf, m, 32); hf += __shfl_xor(hf, m, 32);
    pl += __shfl_xor(pl, m, 32); hl += __shfl_xor(hl, m, 32);
  }
  if (lane == 0) {
    float4 o = {pf, hf, pl, hl};
    *reinterpret_cast<float4*>(posbuf + (size_t)tok * 4) = o;
  }
}

// ---------------------------------------------------------------------------
__global__ void k_cvt(const float* __restrict__ w, unsigned short* __restrict__ wbf) {
  size_t i = ((size_t)blockIdx.x * 256 + threadIdx.x) * 8;
  if (i + 8 <= (size_t)VOCAB * DMODEL) {
    float4 a = *reinterpret_cast<const float4*>(w + i);
    float4 b = *reinterpret_cast<const float4*>(w + i + 4);
    union { unsigned short u[8]; uint4 q; } o;
    o.u[0] = f2bf(a.x); o.u[1] = f2bf(a.y); o.u[2] = f2bf(a.z); o.u[3] = f2bf(a.w);
    o.u[4] = f2bf(b.x); o.u[5] = f2bf(b.y); o.u[6] = f2bf(b.z); o.u[7] = f2bf(b.w);
    *reinterpret_cast<uint4*>(wbf + i) = o.q;
  }
}

// ---------------------------------------------------------------------------
// WMMA scout scan: grid (chunk, vocab-slice). A = 8 scouts x 64 (rows 8..15 = 0).
// ---------------------------------------------------------------------------
__global__ void k_scan_wmma(const float* __restrict__ hidden,
                            const unsigned short* __restrict__ wbf,
                            float* __restrict__ logits_g) {
  __shared__ unsigned short As[16 * SSTRIDE];
  const int c = blockIdx.x, slice = blockIdx.y;
  const int tid = threadIdx.x, wave = tid >> 5, lane = tid & 31;
  const int lhalf = lane >> 4, ncol = lane & 15;

  for (int idx = tid; idx < 16 * LOW; idx += 256) {
    int m = idx >> 6, k = idx & 63;
    float v = (m < 8) ? hidden[(size_t)(c * CHUNK + m * STRIDE) * DMODEL + k] : 0.f;
    As[m * SSTRIDE + k] = f2bf(v);
  }
  __syncthreads();

  const int NT = VOCAB / 16;          // 2000 tiles, 250 per slice
  const int nt0 = slice * (NT / 8), nt1 = nt0 + (NT / 8);
  for (int nt = nt0 + wave; nt < nt1; nt += 8) {
    const unsigned short* brow = wbf + (size_t)(nt * 16 + ncol) * DMODEL;
    v8f acc = {0.f, 0.f, 0.f, 0.f, 0.f, 0.f, 0.f, 0.f};
#pragma unroll
    for (int kb = 0; kb < LOW; kb += 32) {
      union { uint4 q[2]; v16bf v; } bfr;
      bfr.q[0] = *reinterpret_cast<const uint4*>(brow + kb + lhalf * 16);
      bfr.q[1] = *reinterpret_cast<const uint4*>(brow + kb + lhalf * 16 + 8);
      v16bf afr = loadA(As + ncol * SSTRIDE, kb, lhalf);
      acc = __builtin_amdgcn_wmma_f32_16x16x32_bf16(false, afr, false, bfr.v,
                                                    (short)0, acc, false, false);
    }
    if (lhalf == 0) {
      const int v = nt * 16 + ncol;
#pragma unroll
      for (int r = 0; r < 8; r++)
        logits_g[(size_t)(c * NS + r) * VOCAB + v] = acc[r];
    }
  }
}

// ---------------------------------------------------------------------------
// top-256 per scout, deterministic LDS radix select (512 blocks x 256 threads)
// ---------------------------------------------------------------------------
__global__ void k_topk(const float* __restrict__ hidden,
                       const float* __restrict__ w,
                       const float* __restrict__ logits_g,   // nullable
                       int* __restrict__ top_idx) {
  extern __shared__ char smem_raw[];
  float* logits = reinterpret_cast<float*>(smem_raw);     // VOCAB floats
  __shared__ float    scout[LOW];
  __shared__ unsigned hist[256];
  __shared__ unsigned selbin_s, kk_s;
  __shared__ unsigned cntG[256], cntE[256], baseS[256], tieS[256];

  const int c = blockIdx.x >> 3;
  const int s = blockIdx.x & 7;
  const int tid  = threadIdx.x;
  const int wave = tid >> 5, lane = tid & 31;

  if (logits_g) {
    const float* src = logits_g + (size_t)(c * NS + s) * VOCAB;
    for (int i = tid; i < VOCAB; i += 256) logits[i] = src[i];
    __syncthreads();
  } else {
    const int tokrow = c * CHUNK + s * STRIDE;
    if (tid < LOW) scout[tid] = hidden[(size_t)tokrow * DMODEL + tid];
    __syncthreads();
    const float sx = scout[lane * 2], sy = scout[lane * 2 + 1];
    for (int r = wave; r < VOCAB; r += 8) {
      const float2 wv = *reinterpret_cast<const float2*>(w + (size_t)r * DMODEL + lane * 2);
      float p = wv.x * sx + wv.y * sy;
      for (int m = 16; m; m >>= 1) p += __shfl_xor(p, m, 32);
      if (lane == 0) logits[r] = p;
    }
    __syncthreads();
  }

  unsigned prefix = 0, prefmask = 0, kk = KPER;
  for (int pass = 3; pass >= 0; pass--) {
    const int shift = pass * 8;
    hist[tid] = 0;
    __syncthreads();
    for (int i = tid; i < VOCAB; i += 256) {
      unsigned key = f2key(logits[i]);
      if ((key & prefmask) == prefix) atomicAdd(&hist[(key >> shift) & 255u], 1u);
    }
    __syncthreads();
    if (tid == 0) {
      unsigned cnt = 0, b = 0;
      for (int bb = 255; bb >= 0; bb--) {
        if (cnt + hist[bb] >= kk) { b = (unsigned)bb; break; }
        cnt += hist[bb];
      }
      selbin_s = b; kk_s = kk - cnt;
    }
    __syncthreads();
    prefix   |= (selbin_s << shift);
    prefmask |= (255u << shift);
    kk = kk_s;
    __syncthreads();
  }
  const unsigned T = prefix;

  const int per = VOCAB / 256;        // 125
  const int i0 = tid * per, i1 = i0 + per;
  unsigned g = 0, e = 0;
  for (int i = i0; i < i1; i++) {
    unsigned key = f2key(logits[i]);
    if (key > T) g++;
    else if (key == T) e++;
  }
  cntG[tid] = g; cntE[tid] = e;
  __syncthreads();
  if (tid == 0) {
    unsigned run = 0, ties = kk;
    for (int t = 0; t < 256; t++) {
      unsigned ta = cntE[t] < ties ? cntE[t] : ties;
      baseS[t] = run; tieS[t] = ta;
      run += cntG[t] + ta;
      ties -= ta;
    }
  }
  __syncthreads();
  unsigned slot = baseS[tid], tq = tieS[tid];
  int* out = top_idx + (size_t)c * MCAND + s * KPER;
  for (int i = i0; i < i1; i++) {
    unsigned key = f2key(logits[i]);
    bool take = (key > T);
    if (!take && key == T && tq) { take = true; tq--; }
    if (take && slot < KPER) out[slot++] = i;
  }
}

// ---------------------------------------------------------------------------
// GEMM + plain-sum exp accumulation. Grid (chunk, split). Each wave handles
// 4 local tiles (2 groups of 2 N-tiles). Logits are tiny (|v| < ~1) so
// sum-of-exp without max shift is numerically safe; partial sums compose
// linearly across splits.
// ---------------------------------------------------------------------------
__global__ void k_loss(const float* __restrict__ hidden,
                       const float* __restrict__ w,
                       const unsigned short* __restrict__ wbf,   // nullable
                       const int* __restrict__ tgt,
                       const int* __restrict__ top_idx,
                       float* __restrict__ psum,                 // [CCH*NSPLIT*CHUNK]
                       int kdim) {
  extern __shared__ char smem_raw[];
  unsigned short* Asm = reinterpret_cast<unsigned short*>(smem_raw); // [CHUNK][ASTRIDE]
  __shared__ int   cands[LTILES * 16];
  __shared__ int   tloc[CHUNK];
  __shared__ float wsum[8][CHUNK];

  const int c = blockIdx.x, split = blockIdx.y;
  const int tid  = threadIdx.x;
  const int wave = tid >> 5, lane = tid & 31;
  const int lhalf = lane >> 4;
  const int ncol  = lane & 15;

  for (int j = tid; j < LTILES * 16; j += 256)
    cands[j] = top_idx[(size_t)c * MCAND + split * (LTILES * 16) + j];
  if (tid < CHUNK) tloc[tid] = tgt[c * CHUNK + tid];
  for (int idx = tid; idx < CHUNK * kdim; idx += 256) {
    int m = idx / kdim, k = idx - m * kdim;
    Asm[m * ASTRIDE + k] = f2bf(hidden[(size_t)(c * CHUNK + m) * DMODEL + k]);
  }
  __syncthreads();

  float RS[16];
#pragma unroll
  for (int i = 0; i < 16; i++) RS[i] = 0.f;

#pragma unroll
  for (int g = 0; g < 2; g++) {
    const int lt0  = g * 16 + wave * 2;       // local tiles lt0, lt0+1
    const int cid0 = cands[lt0 * 16 + ncol];
    const int cid1 = cands[(lt0 + 1) * 16 + ncol];
    if (g == 0 && wbf) {                       // prefetch next group's rows
      __builtin_prefetch(wbf + (size_t)cands[(16 + wave * 2) * 16 + ncol] * DMODEL, 0, 0);
      __builtin_prefetch(wbf + (size_t)cands[(17 + wave * 2) * 16 + ncol] * DMODEL, 0, 0);
    }
    v8f a00 = {0.f,0.f,0.f,0.f,0.f,0.f,0.f,0.f};
    v8f a10 = a00, a01 = a00, a11 = a00;

#pragma unroll 2
    for (int kb = 0; kb < kdim; kb += 32) {
      const int off = kb + lhalf * 16;
      v16bf b0 = loadB(wbf, w, cid0, off);
      v16bf b1 = loadB(wbf, w, cid1, off);
      v16bf fa0 = loadA(Asm + ncol * ASTRIDE, kb, lhalf);
      v16bf fa1 = loadA(Asm + (16 + ncol) * ASTRIDE, kb, lhalf);
      a00 = __builtin_amdgcn_wmma_f32_16x16x32_bf16(false, fa0, false, b0, (short)0, a00, false, false);
      a10 = __builtin_amdgcn_wmma_f32_16x16x32_bf16(false, fa1, false, b0, (short)0, a10, false, false);
      a01 = __builtin_amdgcn_wmma_f32_16x16x32_bf16(false, fa0, false, b1, (short)0, a01, false, false);
      a11 = __builtin_amdgcn_wmma_f32_16x16x32_bf16(false, fa1, false, b1, (short)0, a11, false, false);
    }

    // mask cand==target and accumulate exp per token (no cross-lane traffic)
#pragma unroll
    for (int t = 0; t < 2; t++) {
      const int cid = t ? cid1 : cid0;
#pragma unroll
      for (int mt = 0; mt < 2; mt++) {
        v8f acc = t ? (mt ? a11 : a01) : (mt ? a10 : a00);
#pragma unroll
        for (int r = 0; r < 8; r++) {
          const int m = mt * 16 + lhalf * 8 + r;
          float e = (cid == tloc[m]) ? 0.f : __expf(acc[r]);
          RS[mt * 8 + r] += e;
        }
      }
    }
  }

  // one cross-lane butterfly per block over the 16 column lanes
#pragma unroll
  for (int i = 0; i < 16; i++)
    for (int mm = 1; mm < 16; mm <<= 1) RS[i] += __shfl_xor(RS[i], mm, 32);

  if (ncol == 0) {
#pragma unroll
    for (int mt = 0; mt < 2; mt++)
#pragma unroll
      for (int r = 0; r < 8; r++)
        wsum[wave][mt * 16 + lhalf * 8 + r] = RS[mt * 8 + r];
  }
  __syncthreads();

  if (tid < CHUNK) {
    float S = 0.f;
    for (int wv = 0; wv < 8; wv++) S += wsum[wv][tid];
    psum[(size_t)(c * NSPLIT + split) * CHUNK + tid] = S;
  }
}

// ---------------------------------------------------------------------------
// finalize: merge splits, add pos + ghost, log, deterministic serial combine
// ---------------------------------------------------------------------------
__global__ void k_finalize(const float* __restrict__ ws_f,
                           const float* __restrict__ posbuf,
                           const float* __restrict__ psum_m,
                           const float* __restrict__ psum_a,
                           float* __restrict__ out) {
  __shared__ float part[256];
  const int tid = threadIdx.x;
  const float wnf = ws_f[0] / (float)VOCAB;
  const float wnl = ws_f[1] / (float)VOCAB;
  const float lvr = logf((float)VREM);
  float local = 0.f;
  for (int tok = tid * 8; tok < tid * 8 + 8; tok++) {
    const int c = tok >> 5, m = tok & 31;
    float Sm = 0.f, Sa = 0.f;
    for (int sp = 0; sp < NSPLIT; sp++) {
      Sm += psum_m[(size_t)(c * NSPLIT + sp) * CHUNK + m];
      Sa += psum_a[(size_t)(c * NSPLIT + sp) * CHUNK + m];
    }
    const float pf = posbuf[tok * 4 + 0], hf = posbuf[tok * 4 + 1];
    const float pl = posbuf[tok * 4 + 2], hl = posbuf[tok * 4 + 3];
    const float gf = lvr + 0.5f * hf * wnf / (float)DMODEL;
    const float ga = lvr + 0.5f * hl * wnl / (float)LOW;
    const float lm = logf(__expf(pf) + Sm + __expf(gf)) - pf;
    const float la = logf(__expf(pl) + Sa + __expf(ga)) - pl;
    local += lm + 0.2f * la;
  }
  part[tid] = local;
  __syncthreads();
  if (tid == 0) {
    float t = 0.f;
    for (int i = 0; i < 256; i++) t += part[i];
    out[0] = t / (float)NTOK;
  }
}

// ---------------------------------------------------------------------------
// ws layout (bytes):
//   [0..8)          : norm sums    [16..1040): norm partials
//   [4096..528384)  : top_idx int[CCH*MCAND]
//   [532480..565248): posbuf  float[NTOK*4]
//   [565248..630784): psum    float[2*CCH*NSPLIT*CHUNK]
//   fast path only:
//   [1MiB  .. +62.5MiB) : bf16 embed table
//   [70MiB .. +62.5MiB) : scout logits f32 (512 x 32000)
// ---------------------------------------------------------------------------
extern "C" void kernel_launch(void* const* d_in, const int* in_sizes, int n_in,
                              void* d_out, int out_size, void* d_ws, size_t ws_size,
                              hipStream_t stream) {
  (void)in_sizes; (void)n_in; (void)out_size;
  const float* hidden = (const float*)d_in[0];
  const float* w      = (const float*)d_in[1];
  const int*   tgt    = (const int*)d_in[2];
  float* ws_f    = (float*)d_ws;
  int*   top_idx = (int*)((char*)d_ws + 4096);
  float* posbuf  = (float*)((char*)d_ws + 532480);
  float* psum_m  = (float*)((char*)d_ws + 565248);
  float* psum_a  = psum_m + (size_t)CCH * NSPLIT * CHUNK;
  float* out     = (float*)d_out;

  const size_t TBL_OFF = (size_t)1 << 20;
  const size_t LOG_OFF = (size_t)70 << 20;
  const size_t NEED    = LOG_OFF + (size_t)CCH * NS * VOCAB * sizeof(float);
  const bool   fast    = (ws_size >= NEED);
  unsigned short* wbf      = fast ? (unsigned short*)((char*)d_ws + TBL_OFF) : (unsigned short*)0;
  float*          logits_g = fast ? (float*)((char*)d_ws + LOG_OFF)          : (float*)0;

  k_norms<<<128, 256, 0, stream>>>(w, ws_f);
  k_norm_combine<<<1, 32, 0, stream>>>(ws_f);
  k_pos<<<NTOK / 8, 256, 0, stream>>>(hidden, w, tgt, posbuf);

  if (fast) {
    k_cvt<<<(VOCAB * DMODEL / 8) / 256, 256, 0, stream>>>(w, wbf);
    k_scan_wmma<<<dim3(CCH, 8), 256, 0, stream>>>(hidden, wbf, logits_g);
  }

  const size_t scanLds = (size_t)VOCAB * sizeof(float);                  // 128 KB
  k_topk<<<CCH * NS, 256, scanLds, stream>>>(hidden, w, logits_g, top_idx);

  const size_t aLds = (size_t)CHUNK * ASTRIDE * sizeof(unsigned short);  // ~66 KB
  k_loss<<<dim3(CCH, NSPLIT), 256, aLds, stream>>>(hidden, w, wbf, tgt, top_idx,
                                                   psum_m, DMODEL);
  k_loss<<<dim3(CCH, NSPLIT), 256, aLds, stream>>>(hidden, w, wbf, tgt, top_idx,
                                                   psum_a, LOW);

  k_finalize<<<1, 256, 0, stream>>>(ws_f, posbuf, psum_m, psum_a, out);
}
